// GAT_50044958933132
// MI455X (gfx1250) — compile-verified, matrix operand order checked by reading, not run on hardware
//
#include <hip/hip_runtime.h>
#include <hip/hip_bf16.h>
#include <math.h>

// ---------------------------------------------------------------------------
// GAT (3-layer, PyG-style) for gfx1250.
//  - Dense GEMMs: v_wmma_f32_16x16x32_bf16, f32 accumulate, bf16 operands
//    pre-packed so the inner loop is pure b128 loads + WMMA.
//  - Edge softmax/aggregation: f32 atomics (max via int-punning trick).
// ---------------------------------------------------------------------------

typedef __attribute__((ext_vector_type(16))) __bf16 v16bf;
typedef __attribute__((ext_vector_type(8)))  __bf16 v8bf;
typedef __attribute__((ext_vector_type(8)))  float  v8f;

#define LEAKY 0.2f

__device__ __forceinline__ __bf16 f2bf(float f) {
  unsigned int u = __builtin_bit_cast(unsigned int, f);
  unsigned int r = (u + 0x7FFFu + ((u >> 16) & 1u)) >> 16;  // RNE truncate
  unsigned short s = (unsigned short)r;
  return __builtin_bit_cast(__bf16, s);
}

__device__ __forceinline__ v16bf cat16(v8bf lo, v8bf hi) {
  return __builtin_shufflevector(lo, hi, 0, 1, 2, 3, 4, 5, 6, 7,
                                         8, 9, 10, 11, 12, 13, 14, 15);
}

__device__ __forceinline__ void atomicMaxF(float* addr, float v) {
  if (v >= 0.0f) atomicMax((int*)addr, __float_as_int(v));
  else           atomicMin((unsigned int*)addr, __float_as_uint(v));
}

// ---------------------------------------------------------------------------
// C[M x Nout] = A[M x 128] * B[128 x Nout].
// A: bf16 row-major [M,128]. Bt: bf16 [Npad,128] = B transposed + zero-padded
// (Npad multiple of 128). One wave -> two adjacent 16x16 tiles (32 cols),
// 4 waves/block -> 128 cols/block. M % 16 == 0 (N=50000=16*3125).
// Per lane, a WMMA fragment is two contiguous 16B runs -> b128 loads.
// ---------------------------------------------------------------------------
__global__ void gat_gemm_wmma_bf16(const __bf16* __restrict__ A,
                                   const __bf16* __restrict__ Bt,
                                   float* __restrict__ C,
                                   int M, int Nout, int ldc) {
  constexpr int K = 128;
  const int lane = threadIdx.x & 31;
  const int wave = threadIdx.x >> 5;
  const int m0   = blockIdx.y * 16;
  const int n0   = (blockIdx.x * 4 + wave) * 32;   // two 16-wide tiles
  const int half = lane >> 4;
  const int l15  = lane & 15;

  const __bf16* Ap = A  + (size_t)(m0 + l15) * K;
  const __bf16* B0 = Bt + (size_t)(n0 + l15) * K;
  const __bf16* B1 = Bt + (size_t)(n0 + 16 + l15) * K;

  v8f acc0 = {}, acc1 = {};
#pragma unroll
  for (int k0 = 0; k0 < K; k0 += 32) {
    const int ko = k0 + 8 * half;   // 16B-aligned offsets
    v16bf a  = cat16(*(const v8bf*)(Ap + ko), *(const v8bf*)(Ap + ko + 16));
    v16bf b0 = cat16(*(const v8bf*)(B0 + ko), *(const v8bf*)(B0 + ko + 16));
    v16bf b1 = cat16(*(const v8bf*)(B1 + ko), *(const v8bf*)(B1 + ko + 16));
    acc0 = __builtin_amdgcn_wmma_f32_16x16x32_bf16(false, a, false, b0,
                                                   (short)0, acc0, false, false);
    acc1 = __builtin_amdgcn_wmma_f32_16x16x32_bf16(false, a, false, b1,
                                                   (short)0, acc1, false, false);
  }

  const int row = m0 + half * 8;
  const int col0 = n0 + l15, col1 = n0 + 16 + l15;
  if (col0 < Nout) {
#pragma unroll
    for (int r = 0; r < 8; ++r) C[(size_t)(row + r) * ldc + col0] = acc0[r];
  }
  if (col1 < Nout) {
#pragma unroll
    for (int r = 0; r < 8; ++r) C[(size_t)(row + r) * ldc + col1] = acc1[r];
  }
}

// f32 -> bf16 elementwise (activation prep; bandwidth-trivial)
__global__ void gat_cvt_bf16(const float* __restrict__ in,
                             __bf16* __restrict__ out, size_t count) {
  size_t i = (size_t)blockIdx.x * blockDim.x + threadIdx.x;
  if (i < count) out[i] = f2bf(in[i]);
}

// Pack weight B[K x Nout] (f32) -> Bt[Npad x K] (bf16, transposed, zero pad)
__global__ void gat_pack_bt(const float* __restrict__ B, __bf16* __restrict__ Bt,
                            int Nout, int Npad) {
  constexpr int K = 128;
  int i = blockIdx.x * blockDim.x + threadIdx.x;  // c*K + k
  if (i >= Npad * K) return;
  const int c = i / K, k = i % K;
  Bt[i] = (c < Nout) ? f2bf(B[(size_t)k * Nout + c]) : f2bf(0.0f);
}

// a_src[n,h] = sum_c xw[n, h*cph + c] * attS[h*cph + c]   (same for a_dst)
__global__ void gat_att_scores(const float* __restrict__ xw,
                               const float* __restrict__ attS,
                               const float* __restrict__ attD,
                               float* __restrict__ as, float* __restrict__ ad,
                               int n, int cph, int ld) {
  int i = blockIdx.x * blockDim.x + threadIdx.x;  // node*2 + head
  if (i >= n * 2) return;
  const int node = i >> 1, h = i & 1;
  const float* row = xw + (size_t)node * ld + h * cph;
  const float* ws  = attS + h * cph;
  const float* wd  = attD + h * cph;
  float ss = 0.f, sd = 0.f;
  for (int c = 0; c < cph; ++c) { float v = row[c]; ss += v * ws[c]; sd += v * wd[c]; }
  as[i] = ss;
  ad[i] = sd;
}

__global__ void gat_fill(float* __restrict__ p, float v, size_t count) {
  size_t i = (size_t)blockIdx.x * blockDim.x + threadIdx.x;
  if (i < count) p[i] = v;
}

// pass 1: per-destination running max of leaky(a_src[src]+a_dst[dst])
__global__ void gat_edge_max(const int* __restrict__ ei, int E, int n,
                             const float* __restrict__ as,
                             const float* __restrict__ ad,
                             float* __restrict__ amax) {
  int e = blockIdx.x * blockDim.x + threadIdx.x;
  if (e >= E + n) return;
  const int s = (e < E) ? ei[e]     : (e - E);   // self-loops appended
  const int d = (e < E) ? ei[E + e] : (e - E);
#pragma unroll
  for (int h = 0; h < 2; ++h) {
    float a = as[s * 2 + h] + ad[d * 2 + h];
    a = (a > 0.f) ? a : LEAKY * a;
    atomicMaxF(&amax[d * 2 + h], a);
  }
}

// pass 2: ex = exp(a - amax[dst]); denom[dst] += ex
__global__ void gat_edge_exp(const int* __restrict__ ei, int E, int n,
                             const float* __restrict__ as,
                             const float* __restrict__ ad,
                             const float* __restrict__ amax,
                             float* __restrict__ ex, float* __restrict__ den) {
  int e = blockIdx.x * blockDim.x + threadIdx.x;
  if (e >= E + n) return;
  const int s = (e < E) ? ei[e]     : (e - E);
  const int d = (e < E) ? ei[E + e] : (e - E);
#pragma unroll
  for (int h = 0; h < 2; ++h) {
    float a = as[s * 2 + h] + ad[d * 2 + h];
    a = (a > 0.f) ? a : LEAKY * a;
    float v = expf(a - amax[d * 2 + h]);
    ex[(size_t)e * 2 + h] = v;
    atomicAdd(&den[d * 2 + h], v);
  }
}

// pass 3: agg[dst, :] += alpha * xw[src, :]   (one wave per edge, coalesced)
__global__ void gat_edge_agg(const int* __restrict__ ei, int E, int n,
                             const float* __restrict__ ex,
                             const float* __restrict__ den,
                             const float* __restrict__ xw,
                             float* __restrict__ agg, int cph, int ld) {
  int w    = (blockIdx.x * blockDim.x + threadIdx.x) >> 5;  // edge id
  int lane = threadIdx.x & 31;
  if (w >= E + n) return;
  const int s = (w < E) ? ei[w]     : (w - E);
  const int d = (w < E) ? ei[E + w] : (w - E);
  const float al0 = ex[(size_t)w * 2 + 0] / den[d * 2 + 0];
  const float al1 = ex[(size_t)w * 2 + 1] / den[d * 2 + 1];
  const int tot = 2 * cph;
  for (int i = lane; i < tot; i += 32) {
    const float al = (i < cph) ? al0 : al1;
    atomicAdd(&agg[(size_t)d * ld + i], al * xw[(size_t)s * ld + i]);
  }
}

// h = elu(agg + conv_bias + skip + lin_bias), emitted directly as bf16
// (h is consumed only by the next layer's bf16 GEMMs)
__global__ void gat_finalize_elu(const float* __restrict__ agg,
                                 const float* __restrict__ skip,
                                 const float* __restrict__ cb,
                                 const float* __restrict__ lb,
                                 __bf16* __restrict__ h, int n, int hc) {
  size_t i = (size_t)blockIdx.x * blockDim.x + threadIdx.x;
  if (i >= (size_t)n * hc) return;
  const int c = (int)(i % hc);
  float v = agg[i] + cb[c] + skip[i] + lb[c];
  v = (v > 0.f) ? v : expm1f(v);
  h[i] = f2bf(v);
}

// out = mean_heads(agg3) + c3_b + skip3 + l3_b   (OUT_DIM=121, ld(agg)=242)
__global__ void gat_finalize_out(const float* __restrict__ agg,
                                 const float* __restrict__ skip,
                                 const float* __restrict__ cb,
                                 const float* __restrict__ lb,
                                 float* __restrict__ out, int n) {
  size_t i = (size_t)blockIdx.x * blockDim.x + threadIdx.x;
  if (i >= (size_t)n * 121) return;
  const int node = (int)(i / 121), o = (int)(i % 121);
  float v = 0.5f * (agg[(size_t)node * 242 + o] + agg[(size_t)node * 242 + 121 + o]);
  out[i] = v + cb[o] + skip[i] + lb[o];
}

// ---------------------------------------------------------------------------

extern "C" void kernel_launch(void* const* d_in, const int* in_sizes, int n_in,
                              void* d_out, int out_size, void* d_ws, size_t ws_size,
                              hipStream_t stream) {
  const float* x    = (const float*)d_in[0];
  const int*   ei   = (const int*)  d_in[1];
  const float* c1W  = (const float*)d_in[2];
  const float* c1as = (const float*)d_in[3];
  const float* c1ad = (const float*)d_in[4];
  const float* c1b  = (const float*)d_in[5];
  const float* c2W  = (const float*)d_in[6];
  const float* c2as = (const float*)d_in[7];
  const float* c2ad = (const float*)d_in[8];
  const float* c2b  = (const float*)d_in[9];
  const float* c3W  = (const float*)d_in[10];
  const float* c3as = (const float*)d_in[11];
  const float* c3ad = (const float*)d_in[12];
  const float* c3b  = (const float*)d_in[13];
  const float* l1W  = (const float*)d_in[14];
  const float* l1b  = (const float*)d_in[15];
  const float* l2W  = (const float*)d_in[16];
  const float* l2b  = (const float*)d_in[17];
  const float* l3W  = (const float*)d_in[18];
  const float* l3b  = (const float*)d_in[19];

  const int Nn = in_sizes[0] / 128;   // 50000 (divisible by 16)
  const int Ee = in_sizes[1] / 2;     // 800000
  const int E2 = Ee + Nn;             // edges incl. self loops

  // workspace carve-out (byte offsets; keep 16B alignment per buffer)
  char* base = (char*)d_ws;
  size_t o = 0;
  auto carve = [&](size_t bytes) {
    void* p = base + o;
    o += (bytes + 255) & ~(size_t)255;
    return p;
  };
  float*  xw   = (float*) carve((size_t)Nn * 242 * 4);  // conv GEMM out
  float*  skip = (float*) carve((size_t)Nn * 128 * 4);  // skip-linear out
  float*  agg  = (float*) carve((size_t)Nn * 242 * 4);  // edge aggregation
  float*  asrc = (float*) carve((size_t)Nn * 2 * 4);
  float*  adst = (float*) carve((size_t)Nn * 2 * 4);
  float*  amax = (float*) carve((size_t)Nn * 2 * 4);
  float*  den  = (float*) carve((size_t)Nn * 2 * 4);
  float*  exb  = (float*) carve((size_t)E2 * 2 * 4);
  __bf16* xbf  = (__bf16*)carve((size_t)Nn * 128 * 2);  // bf16 input acts
  __bf16* hbf  = (__bf16*)carve((size_t)Nn * 128 * 2);  // bf16 layer acts
  __bf16* bt   = (__bf16*)carve((size_t)256 * 128 * 2); // packed weights
  (void)ws_size; (void)n_in; (void)out_size;

  const int TB = 256;
  auto blocks = [](size_t cnt, int tb) { return (unsigned)((cnt + tb - 1) / tb); };

  // pack weights + GEMM (stream-ordered, bt reused)
  auto gemm = [&](const __bf16* A, const float* B, float* C, int Nout, int ldc) {
    const int Npad = ((Nout + 127) / 128) * 128;
    gat_pack_bt<<<blocks((size_t)Npad * 128, TB), TB, 0, stream>>>(B, bt, Nout, Npad);
    dim3 grid(Npad / 128, Nn / 16);
    gat_gemm_wmma_bf16<<<grid, dim3(128), 0, stream>>>(A, bt, C, Nn, Nout, ldc);
  };

  // Edge softmax + aggregation for one layer (xw already computed, ld given).
  auto edge_phase = [&](const float* attS, const float* attD, int cph, int ld) {
    gat_att_scores<<<blocks((size_t)Nn * 2, TB), TB, 0, stream>>>(
        xw, attS, attD, asrc, adst, Nn, cph, ld);
    gat_fill<<<blocks((size_t)Nn * 2, TB), TB, 0, stream>>>(amax, -3.0e38f, (size_t)Nn * 2);
    gat_fill<<<blocks((size_t)Nn * 2, TB), TB, 0, stream>>>(den, 0.0f, (size_t)Nn * 2);
    gat_fill<<<blocks((size_t)Nn * ld, TB), TB, 0, stream>>>(agg, 0.0f, (size_t)Nn * ld);
    gat_edge_max<<<blocks((size_t)E2, TB), TB, 0, stream>>>(ei, Ee, Nn, asrc, adst, amax);
    gat_edge_exp<<<blocks((size_t)E2, TB), TB, 0, stream>>>(ei, Ee, Nn, asrc, adst, amax, exb, den);
    gat_edge_agg<<<blocks((size_t)E2 * 32, TB), TB, 0, stream>>>(
        ei, Ee, Nn, exb, den, xw, agg, cph, ld);
  };

  // ---- Layer 1: in = x [N,128] ----
  gat_cvt_bf16<<<blocks((size_t)Nn * 128, TB), TB, 0, stream>>>(x, xbf, (size_t)Nn * 128);
  gemm(xbf, c1W, xw,   128, 128);
  gemm(xbf, l1W, skip, 128, 128);
  edge_phase(c1as, c1ad, 64, 128);
  gat_finalize_elu<<<blocks((size_t)Nn * 128, TB), TB, 0, stream>>>(
      agg, skip, c1b, l1b, hbf, Nn, 128);

  // ---- Layer 2: in = hbf [N,128] ----
  gemm(hbf, c2W, xw,   128, 128);
  gemm(hbf, l2W, skip, 128, 128);
  edge_phase(c2as, c2ad, 64, 128);
  gat_finalize_elu<<<blocks((size_t)Nn * 128, TB), TB, 0, stream>>>(
      agg, skip, c2b, l2b, hbf, Nn, 128);

  // ---- Layer 3: conv -> [N,2,121], mean over heads ----
  gemm(hbf, c3W, xw,   242, 242);
  gemm(hbf, l3W, skip, 121, 121);
  edge_phase(c3as, c3ad, 121, 242);
  gat_finalize_out<<<blocks((size_t)Nn * 121, TB), TB, 0, stream>>>(
      agg, skip, c3b, l3b, (float*)d_out, Nn);
}
